// NN_67662914781510
// MI455X (gfx1250) — matrix operand order repaired and checked by reading, compiled, not run on hardware
//
#include <hip/hip_runtime.h>

// CDNA5 / gfx1250: wave32, WMMA 16x16x32 bf16 with f32 accumulation.
typedef __attribute__((ext_vector_type(16))) __bf16 v16bf;
typedef __attribute__((ext_vector_type(8)))  float  v8f;

namespace {
constexpr int kH   = 1024;
constexpr int kW   = 1024;
constexpr int kC   = 16;
constexpr int kHID = 128;
constexpr int kTPX = 128;        // pixels per block along x
constexpr int kNW  = 8;          // waves per block (256 threads)
constexpr int kLX  = kTPX + 2;   // staged x extent incl. halo
constexpr int kPXS = 20;         // padded floats per staged pixel (16 + 4)
constexpr int kNFRAG = 20;       // 16 layer-1 (mt*2+kt) + 4 layer-2 (kt)
// packed-weight image: [frag][chunk(2)][lane(32)][8 bf16] = 20*1024 B
constexpr int kWSBF  = kNFRAG * 512;          // bf16 element count = 10240
// hidden staging stride: 128 bf16 + 8 pad = 272 B (16B aligned, 68-bank step)
constexpr int kHST = kHID + 8;
}

// ---------------------------------------------------------------------------
// Pre-pack: W1 (128x48) and W2 (16x128) f32 -> bf16, laid out exactly as the
// per-lane WMMA A fragments (16x32 MxK, this file's contiguous k order: elems 0..15 = K[hi*16..+15];
// K padded to 64 for layer 1).
// ws element index = frag*512 + chunk*256 + lane*8 + e8   (chunk = elem/8)
// ---------------------------------------------------------------------------
__global__ void pack_weights_kernel(const float* __restrict__ W1,
                                    const float* __restrict__ W2,
                                    __bf16* __restrict__ ws) {
  int i = blockIdx.x * 256 + threadIdx.x;
  if (i >= kWSBF) return;
  int frag  = i >> 9;
  int rem   = i & 511;
  int chunk = rem >> 8;
  int lane  = (rem >> 3) & 31;
  int e8    = rem & 7;
  int e     = chunk * 8 + e8;
  int hi = lane >> 4, ln = lane & 15;
  int koff = hi * 16 + e;
  float v;
  if (frag < 16) {                       // layer 1: A = W1 tile
    int mt = frag >> 1, kt = frag & 1;
    int k  = kt * 32 + koff;
    v = (k < 48) ? W1[(mt * 16 + ln) * 48 + k] : 0.0f;
  } else {                               // layer 2: A = W2 tile
    int kt = frag - 16;
    v = W2[ln * kHID + kt * 32 + koff];
  }
  ws[i] = (__bf16)v;
}

// LDS byte address of a generic pointer to __shared__ (aperture: addr[31:0]).
__device__ __forceinline__ unsigned lds_addr_of(const void* p) {
  return (unsigned)(unsigned long long)p;
}

// gfx1250 async global->LDS copy, 16B per lane (ASYNCcnt-tracked, no VGPRs).
__device__ __forceinline__ void async_load_b128(unsigned lds_dst,
                                                const void* gsrc) {
  asm volatile("global_load_async_to_lds_b128 %0, %1, off"
               :: "v"(lds_dst), "v"((unsigned long long)(uintptr_t)gsrc)
               : "memory");
}

__device__ __forceinline__ void wait_async_done() {
  asm volatile("s_wait_asynccnt 0x0" ::: "memory");
}

// Load one A fragment (32 B per lane) from the packed LDS weight image.
__device__ __forceinline__ v16bf load_frag(const __bf16* lw, int frag, int lane) {
  union { uint4 q[2]; v16bf v; } u;
  const char* base = (const char*)lw + frag * 1024 + lane * 16;
  u.q[0] = *(const uint4*)(base);
  u.q[1] = *(const uint4*)(base + 512);
  return u.v;
}

// 1-instruction ReLU: v_med3_f32(0, FLT_MAX, x)
__device__ __forceinline__ float relu1(float x) {
  return __builtin_amdgcn_fmed3f(x, 0.0f, 3.402823466e+38f);
}

__global__ __launch_bounds__(256)
void nca_wmma_kernel(const float* __restrict__ state,
                     const __bf16* __restrict__ ws,
                     const float* __restrict__ b1,
                     const float* __restrict__ b2,
                     float* __restrict__ out)
{
  // Packed bf16 weight fragments (20,480 B), resident all kernel.
  __shared__ __bf16 lw[kWSBF];
  // Arena shared by the (temporally disjoint) halo stage and hidden stage:
  //   phase 1: float ls[3][130][20]            = 31,200 B
  //   phase 2: __bf16 lhid[128 px][136]        = 34,816 B
  __shared__ __align__(16) char smem[kNW * 16 * kHST * 2];
  float*  ls   = (float*)smem;
  __bf16* lhid = (__bf16*)smem;

  const int tid  = threadIdx.x;
  const int wave = tid >> 5;
  const int lane = tid & 31;
  const int hi   = lane >> 4;     // lane half: selects K half / M half
  const int ln   = lane & 15;
  const int y    = blockIdx.y;
  const int x0   = blockIdx.x * kTPX;

  // ---- async copy of packed weights straight into LDS (1280 x b128) ----
  for (int i = tid; i < kWSBF / 8; i += 256)
    async_load_b128(lds_addr_of(&lw[i * 8]), (const uint4*)ws + i);

  // ---- async stage 3 circular rows, pixel-major padded, b128 granules ----
  for (int r = 0; r < 3; ++r) {
    int sy = y + r - 1; sy = (sy < 0) ? kH - 1 : (sy >= kH ? 0 : sy);
    const float* srow = state + (size_t)sy * kW * kC;
    for (int j = tid; j < kLX * 4; j += 256) {   // 520 float4 per row
      int xx = j >> 2, cq = j & 3;
      int sx = x0 + xx - 1; sx = (sx < 0) ? kW - 1 : (sx >= kW ? 0 : sx);
      async_load_b128(lds_addr_of(&ls[(r * kLX + xx) * kPXS + cq * 4]),
                      srow + sx * kC + cq * 4);
    }
  }
  wait_async_done();     // this wave's ASYNCcnt -> 0
  __syncthreads();       // all waves' LDS writes visible

  const int px = wave * 16 + ln;   // this lane's pixel (GEMM column N)
  const int xi = px + 1;

  // ---- perception -> WMMA B fragments, built per-lane ----
  // B(32x16): lanes 0-15 hold K=0..15, lanes 16-31 hold K=16..31, elem e = K.
  // perc K = [gx(0..15) | gy(0..15)] then [x(0..15) | zero-pad].
  // lo lanes need gx, hi lanes gy; both = (p1-tl) + 2*(p2-p3) + (br-p4).
  v16bf bk0, bk1;
  #pragma unroll
  for (int cq = 0; cq < 4; ++cq) {
    #define LSQ(r, x) (*(const float4*)&ls[((r) * kLX + (x)) * kPXS + cq * 4])
    float4 tl4 = LSQ(0, xi - 1), tm4 = LSQ(0, xi), tr4 = LSQ(0, xi + 1);
    float4 ml4 = LSQ(1, xi - 1), mc4 = LSQ(1, xi), mr4 = LSQ(1, xi + 1);
    float4 bl4 = LSQ(2, xi - 1), bm4 = LSQ(2, xi), br4 = LSQ(2, xi + 1);
    #undef LSQ
    const float* tl = (const float*)&tl4; const float* tm = (const float*)&tm4;
    const float* tr = (const float*)&tr4; const float* ml = (const float*)&ml4;
    const float* mc = (const float*)&mc4; const float* mr = (const float*)&mr4;
    const float* bl = (const float*)&bl4; const float* bm = (const float*)&bm4;
    const float* br = (const float*)&br4;
    #pragma unroll
    for (int j = 0; j < 4; ++j) {
      int c = cq * 4 + j;
      float p1 = hi ? bl[j] : tr[j];
      float p2 = hi ? bm[j] : mr[j];
      float p3 = hi ? tm[j] : ml[j];
      float p4 = hi ? tr[j] : bl[j];
      float g  = (p1 - tl[j]) + 2.0f * (p2 - p3) + (br[j] - p4);
      bk0[c] = (__bf16)g;                       // gx (lo) / gy (hi)
      bk1[c] = (__bf16)(hi ? 0.0f : mc[j]);     // x  (lo) / pad (hi)
    }
  }
  __syncthreads();   // all ls reads done; arena becomes lhid

  // ---- layer 1: hidden(128 x 16px) = W1 * perc, bias preloaded into C ----
  v8f acc[8];
  #pragma unroll
  for (int mt = 0; mt < 8; ++mt) {
    const float4* bp = (const float4*)(b1 + mt * 16 + hi * 8);
    float4 bq0 = bp[0], bq1 = bp[1];
    v8f cacc;
    cacc[0] = bq0.x; cacc[1] = bq0.y; cacc[2] = bq0.z; cacc[3] = bq0.w;
    cacc[4] = bq1.x; cacc[5] = bq1.y; cacc[6] = bq1.z; cacc[7] = bq1.w;
    v16bf a0 = load_frag(lw, mt * 2 + 0, lane);
    cacc = __builtin_amdgcn_wmma_f32_16x16x32_bf16(false, a0, false, bk0,
                                                   (short)0, cacc, false, false);
    v16bf a1 = load_frag(lw, mt * 2 + 1, lane);
    cacc = __builtin_amdgcn_wmma_f32_16x16x32_bf16(false, a1, false, bk1,
                                                   (short)0, cacc, false, false);
    acc[mt] = cacc;
  }

  // ---- ReLU (v_med3) + bf16, one ds_store_b128 per M-tile ----
  // D-layout: lane holds M = mt*16 + r + 8*hi of pixel ln -> 8 contiguous bf16.
  #pragma unroll
  for (int mt = 0; mt < 8; ++mt) {
    union { uint4 q; __bf16 h[8]; } s;
    #pragma unroll
    for (int r = 0; r < 8; ++r)
      s.h[r] = (__bf16)relu1(acc[mt][r]);
    *(uint4*)&lhid[(size_t)px * kHST + mt * 16 + hi * 8] = s.q;
  }
  __syncthreads();

  // ---- layer 2: delta(16 x 16px) = W2 * hidden, bias in C ----
  const float4* b2p = (const float4*)(b2 + hi * 8);
  float4 dq0 = b2p[0], dq1 = b2p[1];
  v8f d;
  d[0] = dq0.x; d[1] = dq0.y; d[2] = dq0.z; d[3] = dq0.w;
  d[4] = dq1.x; d[5] = dq1.y; d[6] = dq1.z; d[7] = dq1.w;

  #pragma unroll
  for (int kt = 0; kt < 4; ++kt) {
    v16bf a = load_frag(lw, 16 + kt, lane);
    // B fragment: column = ln, K = kt*32 + hi*16 + e -> 32 B contiguous.
    union { uint4 q[2]; v16bf v; } u;
    const uint4* hp = (const uint4*)&lhid[(size_t)px * kHST + kt * 32 + hi * 16];
    u.q[0] = hp[0];
    u.q[1] = hp[1];
    d = __builtin_amdgcn_wmma_f32_16x16x32_bf16(false, a, false, u.v,
                                                (short)0, d, false, false);
  }

  // ---- store: lane owns pixel (y, x0+px), channels hi*8 .. hi*8+7 ----
  float* op = out + (((size_t)y * kW + (size_t)(x0 + px)) * kC + hi * 8);
  ((float4*)op)[0] = make_float4(d[0], d[1], d[2], d[3]);
  ((float4*)op)[1] = make_float4(d[4], d[5], d[6], d[7]);
}

extern "C" void kernel_launch(void* const* d_in, const int* in_sizes, int n_in,
                              void* d_out, int out_size, void* d_ws, size_t ws_size,
                              hipStream_t stream) {
  (void)in_sizes; (void)n_in; (void)out_size; (void)ws_size;
  const float* state = (const float*)d_in[0];
  const float* W1    = (const float*)d_in[1];
  const float* b1    = (const float*)d_in[2];
  const float* W2    = (const float*)d_in[3];
  const float* b2    = (const float*)d_in[4];
  float* out   = (float*)d_out;
  __bf16* wsbf = (__bf16*)d_ws;

  // 1) pack weights into WMMA fragment order (bf16) in workspace
  pack_weights_kernel<<<(kWSBF + 255) / 256, 256, 0, stream>>>(W1, W2, wsbf);
  // 2) main fused NCA step
  dim3 grid(kW / kTPX, kH);   // (8, 1024) blocks of 256 threads (8 waves)
  nca_wmma_kernel<<<grid, 256, 0, stream>>>(state, wsbf, b1, b2, out);
}
